// HeterogeneousConv_ori_45973329936728
// MI455X (gfx1250) — compile-verified, matrix operand order checked by reading, not run on hardware
//
#include <hip/hip_runtime.h>
#include <hip/hip_bf16.h>
#include <math.h>

// ---------------- problem constants ----------------
#define NB    2
#define CCH   256
#define HH    96
#define WW    96
#define LL    (HH * WW)        // 9216
#define MM    (NB * LL)        // 18432
#define GG    8
#define CG    32
#define OMW   216              // valid offset/mask columns
#define NPAD1 480              // 256 (v) + 216 (om) padded to 16*30
#define KTILES 8               // K=256 / 32
#define MTILES (MM / 16)       // 1152
#define FRAG   512             // bf16 elements per 16x32 fragment tile

typedef __bf16 bf16_t;
typedef __attribute__((ext_vector_type(16))) __bf16 v16bf;
typedef __attribute__((ext_vector_type(8)))  float  v8f;

// ---------------- bf16 split helpers ----------------
__device__ __forceinline__ float bf2f(bf16_t h) {
    unsigned short s = __builtin_bit_cast(unsigned short, h);
    unsigned u = ((unsigned)s) << 16;
    return __builtin_bit_cast(float, u);
}
__device__ __forceinline__ bf16_t f2bf(float f) {   // round-to-nearest-even
    unsigned u = __builtin_bit_cast(unsigned, f);
    unsigned r = (u + 0x7FFFu + ((u >> 16) & 1u)) >> 16;
    unsigned short s = (unsigned short)r;
    return __builtin_bit_cast(bf16_t, s);
}
__device__ __forceinline__ void split_bf16(float f, bf16_t& hi, bf16_t& lo) {
    hi = f2bf(f);
    lo = f2bf(f - bf2f(hi));
}

// ---------------- fragment index maps (ISA 16-bit A 16x32 layout; B = mirror) ----------------
__device__ __forceinline__ size_t amap(int m, int k) {
    int mt = m >> 4, m16 = m & 15, kt = k >> 5, k32 = k & 31;
    int hib = k32 >> 4, w16 = k32 & 15, lg = w16 >> 3, w8 = w16 & 7;
    int vg = hib * 4 + (w8 >> 1), hf = w8 & 1;
    int lane = lg * 16 + m16, hidx = vg * 2 + hf;
    return ((size_t)(mt * KTILES + kt)) * FRAG + (size_t)lane * 16 + hidx;
}
__device__ __forceinline__ size_t bmap(int k, int n) {
    int nt = n >> 4, n16 = n & 15, kt = k >> 5, k32 = k & 31;
    int lg = (k32 & 15) >> 3;
    int vg = (k32 >> 4) * 4 + ((k32 & 7) >> 1), hf = k32 & 1;
    int lane = lg * 16 + n16, hidx = vg * 2 + hf;
    return ((size_t)(nt * KTILES + kt)) * FRAG + (size_t)lane * 16 + hidx;
}

// ---------------- pack x: NCHW -> A fragments (hi/lo bf16) ----------------
__global__ void pack_x_kernel(const float* __restrict__ x,
                              bf16_t* __restrict__ Ahi, bf16_t* __restrict__ Alo) {
    int m = blockIdx.x * 256 + threadIdx.x;      // token index (n*L + l)
    int k = blockIdx.y;                          // channel
    int n = m / LL, l = m - n * LL;
    float v = x[((size_t)n * CCH + k) * LL + l]; // coalesced in l
    bf16_t hi, lo; split_bf16(v, hi, lo);
    size_t d = amap(m, k);
    Ahi[d] = hi; Alo[d] = lo;
}

// ---------------- pack weights: y = x @ W.T  =>  B[k][n] = W[n][k] ----------------
__global__ void pack_w_kernel(const float* __restrict__ W0, const float* __restrict__ W1,
                              int nv0, int nv1, int npad,
                              bf16_t* __restrict__ Bhi, bf16_t* __restrict__ Blo) {
    int tid = blockIdx.x * 256 + threadIdx.x;
    if (tid >= npad * CCH) return;
    int k = tid / npad, n = tid - k * npad;
    float v = 0.f;
    if (n < nv0)            v = W0[(size_t)n * CCH + k];
    else if (n < nv0 + nv1) v = W1[(size_t)(n - nv0) * CCH + k];
    bf16_t hi, lo; split_bf16(v, hi, lo);
    size_t d = bmap(k, n);
    Bhi[d] = hi; Blo[d] = lo;
}

__global__ void pack_bias1_kernel(const float* __restrict__ vb, const float* __restrict__ ob,
                                  float* __restrict__ bias1) {
    int n = blockIdx.x * 256 + threadIdx.x;
    if (n >= NPAD1) return;
    bias1[n] = (n < CCH) ? vb[n] : ((n < CCH + OMW) ? ob[n - CCH] : 0.f);
}

// ---------------- WMMA GEMM: 32x32 output per wave (2x2 register tile), K=256 ----------------
// 3xBF16 split product per tile pair: hi*hi + hi*lo + lo*hi  -> 12 WMMAs per K-step.
// mode 0: out[m*ldc + col]  (fused v||om buffer)
// mode 1: out[(n*C + col)*L + l]  (final NCHW scatter)
__global__ void gemm_bf16x3_kernel(const bf16_t* __restrict__ Ahi, const bf16_t* __restrict__ Alo,
                                   const bf16_t* __restrict__ Bhi, const bf16_t* __restrict__ Blo,
                                   const float* __restrict__ bias, float* __restrict__ out,
                                   int mode, int ldc) {
    int lane = threadIdx.x & 31;
    int wave = threadIdx.x >> 5;
    int mt0 = (blockIdx.x * 8 + wave) * 2;       // two adjacent M tiles
    int nt0 = blockIdx.y * 2;                    // two adjacent N tiles

    size_t a0 = (size_t)mt0 * KTILES * FRAG + (size_t)lane * 16;
    size_t a1 = a0 + (size_t)KTILES * FRAG;
    size_t b0 = (size_t)nt0 * KTILES * FRAG + (size_t)lane * 16;
    size_t b1 = b0 + (size_t)KTILES * FRAG;

    v8f acc00 = {0.f,0.f,0.f,0.f,0.f,0.f,0.f,0.f};
    v8f acc01 = acc00, acc10 = acc00, acc11 = acc00;

    #pragma unroll
    for (int kt = 0; kt < KTILES; ++kt) {
        size_t ko = (size_t)kt * FRAG;
        v16bf a0h = *(const v16bf*)(Ahi + a0 + ko);
        v16bf a0l = *(const v16bf*)(Alo + a0 + ko);
        v16bf a1h = *(const v16bf*)(Ahi + a1 + ko);
        v16bf a1l = *(const v16bf*)(Alo + a1 + ko);
        v16bf b0h = *(const v16bf*)(Bhi + b0 + ko);
        v16bf b0l = *(const v16bf*)(Blo + b0 + ko);
        v16bf b1h = *(const v16bf*)(Bhi + b1 + ko);
        v16bf b1l = *(const v16bf*)(Blo + b1 + ko);
        if (kt + 1 < KTILES) {   // global_prefetch_b8 for next K-step
            __builtin_prefetch((const void*)(Ahi + a0 + ko + FRAG), 0, 1);
            __builtin_prefetch((const void*)(Ahi + a1 + ko + FRAG), 0, 1);
            __builtin_prefetch((const void*)(Bhi + b0 + ko + FRAG), 0, 1);
            __builtin_prefetch((const void*)(Bhi + b1 + ko + FRAG), 0, 1);
        }
        acc00 = __builtin_amdgcn_wmma_f32_16x16x32_bf16(false, a0h, false, b0h, (short)0, acc00, false, false);
        acc00 = __builtin_amdgcn_wmma_f32_16x16x32_bf16(false, a0h, false, b0l, (short)0, acc00, false, false);
        acc00 = __builtin_amdgcn_wmma_f32_16x16x32_bf16(false, a0l, false, b0h, (short)0, acc00, false, false);

        acc01 = __builtin_amdgcn_wmma_f32_16x16x32_bf16(false, a0h, false, b1h, (short)0, acc01, false, false);
        acc01 = __builtin_amdgcn_wmma_f32_16x16x32_bf16(false, a0h, false, b1l, (short)0, acc01, false, false);
        acc01 = __builtin_amdgcn_wmma_f32_16x16x32_bf16(false, a0l, false, b1h, (short)0, acc01, false, false);

        acc10 = __builtin_amdgcn_wmma_f32_16x16x32_bf16(false, a1h, false, b0h, (short)0, acc10, false, false);
        acc10 = __builtin_amdgcn_wmma_f32_16x16x32_bf16(false, a1h, false, b0l, (short)0, acc10, false, false);
        acc10 = __builtin_amdgcn_wmma_f32_16x16x32_bf16(false, a1l, false, b0h, (short)0, acc10, false, false);

        acc11 = __builtin_amdgcn_wmma_f32_16x16x32_bf16(false, a1h, false, b1h, (short)0, acc11, false, false);
        acc11 = __builtin_amdgcn_wmma_f32_16x16x32_bf16(false, a1h, false, b1l, (short)0, acc11, false, false);
        acc11 = __builtin_amdgcn_wmma_f32_16x16x32_bf16(false, a1l, false, b1h, (short)0, acc11, false, false);
    }

    // C/D layout: lanes 0-15 -> N=lane, M=r; lanes 16-31 -> N=lane-16, M=8+r
    int n16 = lane & 15;
    int lg  = lane >> 4;
    #pragma unroll
    for (int t = 0; t < 4; ++t) {
        int mt = mt0 + (t >> 1);
        int nt = nt0 + (t & 1);
        v8f acc = (t == 0) ? acc00 : (t == 1) ? acc01 : (t == 2) ? acc10 : acc11;
        int col = (nt << 4) + n16;
        float b = bias[col];
        int m0 = (mt << 4) + (lg << 3);
        #pragma unroll
        for (int r = 0; r < 8; ++r) {
            int m = m0 + r;
            float val = acc[r] + b;
            if (mode == 0) {
                out[(size_t)m * ldc + col] = val;
            } else {
                int n = m / LL, l = m - n * LL;
                out[((size_t)n * CCH + col) * LL + l] = val;
            }
        }
    }
}

// ---------------- deformable sampling: one wave per (n,l,g); lane = channel ----------------
__global__ void dcn_sample_kernel(const float* __restrict__ vom,
                                  bf16_t* __restrict__ Shi, bf16_t* __restrict__ Slo) {
    int lane = threadIdx.x & 31;
    int wave = threadIdx.x >> 5;
    int idx = blockIdx.x * 8 + wave;          // (n*L + l)*G + g
    int g  = idx & 7;
    int nl = idx >> 3;
    int n  = nl / LL, l = nl - n * LL;
    int hh = l / WW,  ww = l - hh * WW;

    const float* om = vom + (size_t)nl * NPAD1 + CCH + g * 27;   // 18 offsets + 9 masks
    const float* vb = vom + (size_t)n * LL * NPAD1 + g * CG + lane;

    float acc = 0.f;
    #pragma unroll
    for (int k = 0; k < 9; ++k) {
        float offx = om[2 * k], offy = om[2 * k + 1], mk = om[18 + k];
        float lx = (float)ww + (float)(k % 3 - 1) + offx;
        float ly = (float)hh + (float)(k / 3 - 1) + offy;
        float fx = floorf(lx), fy = floorf(ly);
        int x0 = (int)fx, y0 = (int)fy;
        float wx = lx - fx, wy = ly - fy;

        float s = 0.f;
        #pragma unroll
        for (int c = 0; c < 4; ++c) {
            int yi = y0 + (c >> 1);
            int xi = x0 + (c & 1);
            float wgt = ((c >> 1) ? wy : 1.f - wy) * ((c & 1) ? wx : 1.f - wx);
            bool valid = (xi >= 0) & (xi < WW) & (yi >= 0) & (yi < HH);
            int cy = yi < 0 ? 0 : (yi > HH - 1 ? HH - 1 : yi);
            int cx = xi < 0 ? 0 : (xi > WW - 1 ? WW - 1 : xi);
            float v = vb[(size_t)(cy * WW + cx) * NPAD1];   // coalesced across lanes
            s += wgt * (valid ? v : 0.f);
        }
        acc += mk * s;
    }

    // write directly into packed A fragments for the output-proj GEMM
    int m = nl;                      // row
    int kch = g * CG + lane;         // column (channel)
    bf16_t hi, lo; split_bf16(acc, hi, lo);
    size_t d = amap(m, kch);
    Shi[d] = hi; Slo[d] = lo;
}

// ---------------- workspace layout ----------------
static constexpr size_t SZ_APACK = (size_t)MM * CCH * 2;       // 9,437,184
static constexpr size_t SZ_B1    = (size_t)NPAD1 * CCH * 2;    //   245,760
static constexpr size_t SZ_B2    = (size_t)CCH * CCH * 2;      //   131,072
static constexpr size_t OFF_AHI   = 0;
static constexpr size_t OFF_ALO   = OFF_AHI + SZ_APACK;
static constexpr size_t OFF_B1HI  = OFF_ALO + SZ_APACK;
static constexpr size_t OFF_B1LO  = OFF_B1HI + SZ_B1;
static constexpr size_t OFF_B2HI  = OFF_B1LO + SZ_B1;
static constexpr size_t OFF_B2LO  = OFF_B2HI + SZ_B2;
static constexpr size_t OFF_BIAS1 = OFF_B2LO + SZ_B2;
static constexpr size_t OFF_VOM   = (OFF_BIAS1 + NPAD1 * 4 + 255) & ~(size_t)255;
// vom: MM * NPAD1 * 4 = 35,389,440 bytes;  total ~55 MB.
// sampled fragments reuse the A-fragment region (GEMM1 has finished reading it).

extern "C" void kernel_launch(void* const* d_in, const int* in_sizes, int n_in,
                              void* d_out, int out_size, void* d_ws, size_t ws_size,
                              hipStream_t stream) {
    const float* x   = (const float*)d_in[0];
    const float* vpw = (const float*)d_in[1];
    const float* vpb = (const float*)d_in[2];
    const float* omw = (const float*)d_in[3];
    const float* omb = (const float*)d_in[4];
    const float* opw = (const float*)d_in[5];
    const float* opb = (const float*)d_in[6];
    float* out = (float*)d_out;
    char* ws = (char*)d_ws;

    bf16_t* Ahi  = (bf16_t*)(ws + OFF_AHI);
    bf16_t* Alo  = (bf16_t*)(ws + OFF_ALO);
    bf16_t* B1hi = (bf16_t*)(ws + OFF_B1HI);
    bf16_t* B1lo = (bf16_t*)(ws + OFF_B1LO);
    bf16_t* B2hi = (bf16_t*)(ws + OFF_B2HI);
    bf16_t* B2lo = (bf16_t*)(ws + OFF_B2LO);
    float*  bias1 = (float*)(ws + OFF_BIAS1);
    float*  vom   = (float*)(ws + OFF_VOM);

    // 1) pack weights, biases, activations
    pack_w_kernel<<<dim3((NPAD1 * CCH) / 256), 256, 0, stream>>>(vpw, omw, CCH, OMW, NPAD1, B1hi, B1lo);
    pack_w_kernel<<<dim3((CCH * CCH) / 256), 256, 0, stream>>>(opw, nullptr, CCH, 0, CCH, B2hi, B2lo);
    pack_bias1_kernel<<<dim3(2), 256, 0, stream>>>(vpb, omb, bias1);
    pack_x_kernel<<<dim3(MM / 256, CCH), 256, 0, stream>>>(x, Ahi, Alo);

    // 2) fused value + offset/mask projection: vom[M x 480]
    //    grid: 1152/2 = 576 M-tile pairs / 8 waves = 72 blocks;  480/32 = 15 N pairs
    gemm_bf16x3_kernel<<<dim3(MTILES / 2 / 8, NPAD1 / 32), 256, 0, stream>>>(
        Ahi, Alo, B1hi, B1lo, bias1, vom, /*mode=*/0, /*ldc=*/NPAD1);

    // 3) deformable bilinear sampling -> packed fragments (reuse A region)
    dcn_sample_kernel<<<dim3((MM * GG) / 8), 256, 0, stream>>>(vom, Ahi, Alo);

    // 4) output projection, scattered to NCHW;  256/32 = 8 N pairs
    gemm_bf16x3_kernel<<<dim3(MTILES / 2 / 8, CCH / 32), 256, 0, stream>>>(
        Ahi, Alo, B2hi, B2lo, opb, out, /*mode=*/1, /*ldc=*/0);
}
